// Rnn_86045374808728
// MI455X (gfx1250) — matrix operand order repaired and checked by reading, not compiled, not run on hardware
//
#include <hip/hip_runtime.h>
#include <hip/hip_bf16.h>

typedef __attribute__((ext_vector_type(16))) _Float16 v16h;
typedef __attribute__((ext_vector_type(8)))  _Float16 v8h;
typedef __attribute__((ext_vector_type(2)))  _Float16 v2h;
typedef __attribute__((ext_vector_type(8)))  float    v8f;

#if __has_builtin(__builtin_amdgcn_tanhf)
#define TANHF(x) __builtin_amdgcn_tanhf(x)
#else
#define TANHF(x) tanhf(x)
#endif

#define WMMA_F32_F16(A, Bm, C) \
  __builtin_amdgcn_wmma_f32_16x16x32_f16(false, (A), false, (Bm), (short)0, (C), false, false)

static constexpr int H   = 32;             // hidden size (== WMMA K)
static constexpr int WPB = 4;              // waves per block
static constexpr int NT  = 2;              // batch tiles (independent chains) per wave

// K-permutation: WMMA contraction is invariant under a K-permutation applied to both
// A and B. Chosen so a lane's two post-tanh outputs (hidden lr, lr+16) pack into one
// b32 LDS store (merges to ds_store_2addr_b32) and the A reload is 2x ds_load_b128.
__host__ __device__ static inline int hiddenOfK(int kslot) {
  const int p = (kslot & 7) | ((kslot & 8) << 1) | ((kslot & 16) >> 1);
  return ((p & 1) << 4) | (p >> 1);
}

__global__ __launch_bounds__(32 * WPB)
void rnn2_wmma_kernel(const float* __restrict__ x,      // [B,T,1]
                      const float* __restrict__ h0,     // [2,B,32]
                      const float* __restrict__ W_ih0,  // [32,1]
                      const float* __restrict__ W_hh0,  // [32,32]
                      const float* __restrict__ b_ih0,  // [32]
                      const float* __restrict__ b_hh0,  // [32]
                      const float* __restrict__ W_ih1,  // [32,32]
                      const float* __restrict__ W_hh1,  // [32,32]
                      const float* __restrict__ b_ih1,  // [32]
                      const float* __restrict__ b_hh1,  // [32]
                      const float* __restrict__ W_out,  // [1,32]
                      const float* __restrict__ b_out,  // [1]
                      float* __restrict__ outs,         // [B,T]
                      float* __restrict__ hstate,       // [2,B,32]
                      int B, int T)
{
  __shared__ _Float16 smem[WPB * NT * 2 * 16 * H];  // per wave: 2 tiles x 2 layers

  const int lane = threadIdx.x & 31;
  const int wave = threadIdx.x >> 5;
  const int gw   = blockIdx.x * WPB + wave;    // global wave id
  if (gw * NT * 16 >= B) return;               // wave-uniform

  const int half = lane >> 4;                  // 0: lanes 0-15, 1: lanes 16-31
  const int lr   = lane & 15;

  // Two batch tiles per wave; clamp tile1 onto tile0 at the tail (duplicate,
  // in-bounds, identical writes -> deterministic).
  int bb[NT];
  bb[0] = gw * NT * 16;
  bb[1] = (bb[0] + 2 * 16 <= B) ? (bb[0] + 16) : bb[0];

  _Float16* smBase = &smem[wave * (NT * 2 * 16 * H)];
  _Float16* sm1[NT];
  _Float16* sm2[NT];
#pragma unroll
  for (int ti = 0; ti < NT; ++ti) {
    sm1[ti] = smBase + ti * (2 * 16 * H);
    sm2[ti] = sm1[ti] + 16 * H;
  }

  // ---- Constant B-fragments (Bmat[k][n] = W[n][k]) with permuted K order (shared). ----
  const int kk = hiddenOfK(lr + 16 * half);
  auto buildB = [&](const float* W, int nt) -> v16h {
    v16h b;
#pragma unroll
    for (int j = 0; j < 8; ++j) {
      b[j]     = (_Float16)W[(16 * nt + lr) * H + hiddenOfK(8 * half + j)];
      b[8 + j] = (_Float16)W[(16 * nt + lr) * H + hiddenOfK(16 + 8 * half + j)];
    }
    return b;
  };
  const v16h Bhh0a = buildB(W_hh0, 0), Bhh0b = buildB(W_hh0, 1);
  const v16h Bih1a = buildB(W_ih1, 0), Bih1b = buildB(W_ih1, 1);
  const v16h Bhh1a = buildB(W_hh1, 0), Bhh1b = buildB(W_hh1, 1);

  v16h Bout;                                   // w_out in column N=0, rest zero
#pragma unroll
  for (int e = 0; e < 16; ++e) Bout[e] = (_Float16)0.0f;
  if (lr == 0) {
#pragma unroll
    for (int j = 0; j < 8; ++j) {
      Bout[j]     = (_Float16)W_out[hiddenOfK(8 * half + j)];
      Bout[8 + j] = (_Float16)W_out[hiddenOfK(16 + 8 * half + j)];
    }
  }

  // ---- Per-lane scalars (shared across tiles) ----
  const float bias1a = b_ih0[lr] + b_hh0[lr];
  const float bias1b = b_ih0[16 + lr] + b_hh0[16 + lr];
  const float wxa    = W_ih0[lr];
  const float wxb    = W_ih0[16 + lr];
  const float bias2a = b_ih1[lr] + b_hh1[lr];
  const float bias2b = b_ih1[16 + lr] + b_hh1[16 + lr];
  v8f c2a, c2b, Cy;
#pragma unroll
  for (int r = 0; r < 8; ++r) { c2a[r] = bias2a; c2b[r] = bias2b; }
  {
    const float cy = (lr == 0) ? b_out[0] : 0.0f;
#pragma unroll
    for (int r = 0; r < 8; ++r) Cy[r] = cy;
  }

  // ---- A-fragment helpers (permuted K order) ----
  auto loadA_global = [&](const float* hsrc, int brow) -> v16h {
    v16h a;
#pragma unroll
    for (int j = 0; j < 8; ++j) {
      a[j]     = (_Float16)hsrc[(brow + lr) * H + hiddenOfK(8 * half + j)];
      a[8 + j] = (_Float16)hsrc[(brow + lr) * H + hiddenOfK(16 + 8 * half + j)];
    }
    return a;
  };
  auto loadA_lds = [&](const _Float16* p) -> v16h {
    v8h lo, hi;                                 // two contiguous 16B -> ds_load_b128 x2
    __builtin_memcpy(&lo, p + lr * H + 16 * half, 16);
    __builtin_memcpy(&hi, p + lr * H + 16 * half + 8, 16);
    v16h a;
#pragma unroll
    for (int j = 0; j < 8; ++j) { a[j] = lo[j]; a[8 + j] = hi[j]; }
    return a;
  };
  auto storeT = [&](_Float16* p, int r, float va, float vb) {
    v2h pk;
    pk[0] = (_Float16)va;
    pk[1] = (_Float16)vb;
    __builtin_memcpy(p + (r + 8 * half) * H + 2 * lr, &pk, 4);
  };

  v16h Ah1[NT], Ah2[NT];
#pragma unroll
  for (int ti = 0; ti < NT; ++ti) {
    Ah1[ti] = loadA_global(h0, bb[ti]);
    Ah2[ti] = loadA_global(h0 + (size_t)B * H, bb[ti]);
  }

  // x feed: ONE per-lane load covers all 32 batch rows of both tiles
  // (lanes 0-15 -> tile0 rows, lanes 16-31 -> tile1 rows), then 16 ds_bpermute
  // distribute it; both pipelined one step ahead of use.
  const float* xrow = x + (size_t)(bb[half] + lr) * T;
  float xs[NT][8];
  {
    const float vx0 = xrow[0];
#pragma unroll
    for (int r = 0; r < 8; ++r) {
      xs[0][r] = __shfl(vx0, r + 8 * half, 32);
      xs[1][r] = __shfl(vx0, 16 + r + 8 * half, 32);
    }
  }

  v8f d1a[NT], d1b[NT], d2a[NT], d2b[NT];
#pragma unroll
  for (int ti = 0; ti < NT; ++ti) { d1a[ti] = {}; d1b[ti] = {}; d2a[ti] = {}; d2b[ti] = {}; }

#pragma unroll 1
  for (int t = 0; t < T; ++t) {
    const int tn = (t + 1 < T) ? (t + 1) : t;
    const float vxn = xrow[tn];                // prefetch next step's x (off critical path)

    // ---- Layer 1 pre-activation, both tiles ----
    v8f c1a[NT], c1b[NT];
#pragma unroll
    for (int ti = 0; ti < NT; ++ti)
#pragma unroll
      for (int r = 0; r < 8; ++r) {
        c1a[ti][r] = bias1a + xs[ti][r] * wxa;
        c1b[ti][r] = bias1b + xs[ti][r] * wxb;
      }
#pragma unroll
    for (int ti = 0; ti < NT; ++ti) {
      d1a[ti] = WMMA_F32_F16(Ah1[ti], Bhh0a, c1a[ti]);
      d1b[ti] = WMMA_F32_F16(Ah1[ti], Bhh0b, c1b[ti]);
    }

    // ---- Layer-2 recurrent half (independent of new h1): issue early ----
    v8f e2a[NT], e2b[NT];
#pragma unroll
    for (int ti = 0; ti < NT; ++ti) {
      e2a[ti] = WMMA_F32_F16(Ah2[ti], Bhh1a, c2a);
      e2b[ti] = WMMA_F32_F16(Ah2[ti], Bhh1b, c2b);
    }

#pragma unroll
    for (int ti = 0; ti < NT; ++ti)
#pragma unroll
      for (int r = 0; r < 8; ++r) { d1a[ti][r] = TANHF(d1a[ti][r]); d1b[ti][r] = TANHF(d1b[ti][r]); }
#pragma unroll
    for (int ti = 0; ti < NT; ++ti) {
#pragma unroll
      for (int r = 0; r < 8; ++r) storeT(sm1[ti], r, d1a[ti][r], d1b[ti][r]);
      Ah1[ti] = loadA_lds(sm1[ti]);
    }

    // ---- Layer-2 input half, C-chained onto recurrent half ----
#pragma unroll
    for (int ti = 0; ti < NT; ++ti) {
      d2a[ti] = WMMA_F32_F16(Ah1[ti], Bih1a, e2a[ti]);
      d2b[ti] = WMMA_F32_F16(Ah1[ti], Bih1b, e2b[ti]);
    }
#pragma unroll
    for (int ti = 0; ti < NT; ++ti)
#pragma unroll
      for (int r = 0; r < 8; ++r) { d2a[ti][r] = TANHF(d2a[ti][r]); d2b[ti][r] = TANHF(d2b[ti][r]); }
#pragma unroll
    for (int ti = 0; ti < NT; ++ti) {
#pragma unroll
      for (int r = 0; r < 8; ++r) storeT(sm2[ti], r, d2a[ti][r], d2b[ti][r]);
      Ah2[ti] = loadA_lds(sm2[ti]);
    }

    // ---- Output head via WMMA: y tile column 0 = h2_new @ w_out + b_out ----
    v8f dy[NT];
#pragma unroll
    for (int ti = 0; ti < NT; ++ti) dy[ti] = WMMA_F32_F16(Ah2[ti], Bout, Cy);
    if (lr == 0) {                              // lanes 0 and 16 hold column N=0
#pragma unroll
      for (int ti = 0; ti < NT; ++ti)
#pragma unroll
        for (int r = 0; r < 8; ++r)
          outs[(size_t)(bb[ti] + r + 8 * half) * T + t] = dy[ti][r];
    }

    // ---- Distribute next step's x (vxn has had the whole body to arrive) ----
#pragma unroll
    for (int r = 0; r < 8; ++r) {
      xs[0][r] = __shfl(vxn, r + 8 * half, 32);
      xs[1][r] = __shfl(vxn, 16 + r + 8 * half, 32);
    }
  }

  // ---- Final hidden states (post-tanh f32 straight from last D registers) ----
#pragma unroll
  for (int ti = 0; ti < NT; ++ti)
#pragma unroll
    for (int r = 0; r < 8; ++r) {
      const int row = bb[ti] + r + 8 * half;
      hstate[(size_t)row * H + lr]            = d1a[ti][r];
      hstate[(size_t)row * H + 16 + lr]       = d1b[ti][r];
      hstate[(size_t)(B + row) * H + lr]      = d2a[ti][r];
      hstate[(size_t)(B + row) * H + 16 + lr] = d2b[ti][r];
    }
}

extern "C" void kernel_launch(void* const* d_in, const int* in_sizes, int n_in,
                              void* d_out, int out_size, void* d_ws, size_t ws_size,
                              hipStream_t stream) {
  const float* x     = (const float*)d_in[0];
  const float* h0    = (const float*)d_in[1];
  const float* W_ih0 = (const float*)d_in[2];
  const float* W_hh0 = (const float*)d_in[3];
  const float* b_ih0 = (const float*)d_in[4];
  const float* b_hh0 = (const float*)d_in[5];
  const float* W_ih1 = (const float*)d_in[6];
  const float* W_hh1 = (const float*)d_in[7];
  const float* b_ih1 = (const float*)d_in[8];
  const float* b_hh1 = (const float*)d_in[9];
  const float* W_out = (const float*)d_in[10];
  const float* b_out = (const float*)d_in[11];

  const int B = in_sizes[1] / (2 * H);          // h0 is [2,B,32]
  const int T = (B > 0) ? in_sizes[0] / B : 0;  // x is [B,T,1]

  float* outs   = (float*)d_out;                // [B,T] flat
  float* hstate = outs + (size_t)B * T;         // [2,B,32] flat

  const int waves  = (B + NT * 16 - 1) / (NT * 16);
  const int blocks = (waves + WPB - 1) / WPB;
  rnn2_wmma_kernel<<<blocks, 32 * WPB, 0, stream>>>(
      x, h0, W_ih0, W_hh0, b_ih0, b_hh0, W_ih1, W_hh1, b_ih1, b_hh1,
      W_out, b_out, outs, hstate, B, T);
}
